// BNBQuantizedLinear_81381040325021
// MI455X (gfx1250) — matrix operand order repaired and checked by reading, not compiled, not run on hardware
//
#include <hip/hip_runtime.h>

// ---------------------------------------------------------------------------
// BNB group-quantized linear for MI455X (gfx1250, wave32, WMMA)
//   out[8192,11008] = x[8192,4096] @ dequant(W[11008,4096])^T + bias
// Pass 1: per-group (scale, min) -> d_ws  (group = 128 contiguous K elems)
// Pass 2: tiled bf16 WMMA GEMM with fused dequant (w*scale + min)
//         block tile 256x128, 8 waves, 64x64 per wave (16 WMMA / K-substep)
// ---------------------------------------------------------------------------

typedef __attribute__((ext_vector_type(16))) __bf16 v16bf;
typedef __attribute__((ext_vector_type(8)))  __bf16 v8bf;
typedef __attribute__((ext_vector_type(8)))  float  v8f;

#define M_DIM 8192
#define K_DIM 4096
#define N_DIM 11008
#define QGROUP 128
#define GROUPS_PER_ROW (K_DIM / QGROUP)        // 32
#define NGROUPS (N_DIM * GROUPS_PER_ROW)       // 352256
#define BM 256
#define BN 128
#define BK 64
#define LDS_STRIDE (BK + 8)                    // 72 bf16 = 144 B (pad: no bank conflicts)

__device__ __forceinline__ __bf16 f2bf(float f) {
    // round-to-nearest-even fp32 -> bf16
    unsigned int u = __float_as_uint(f);
    u += 0x7FFFu + ((u >> 16) & 1u);
    unsigned short s = (unsigned short)(u >> 16);
    __bf16 r;
    __builtin_memcpy(&r, &s, sizeof(r));
    return r;
}

// ---------------- Pass 1: per-group scale/min (one wave32 per group) -------
__global__ __launch_bounds__(256)
void bnb_group_stats(const float* __restrict__ W, float2* __restrict__ sm) {
    const int lane = threadIdx.x & 31;
    const int g = (int)((blockIdx.x * blockDim.x + threadIdx.x) >> 5);
    if (g >= NGROUPS) return;
    const float4 v = *(const float4*)(W + (size_t)g * QGROUP + lane * 4);
    float mn = fminf(fminf(v.x, v.y), fminf(v.z, v.w));
    float mx = fmaxf(fmaxf(v.x, v.y), fmaxf(v.z, v.w));
#pragma unroll
    for (int m = 16; m >= 1; m >>= 1) {
        mn = fminf(mn, __shfl_xor(mn, m, 32));
        mx = fmaxf(mx, __shfl_xor(mx, m, 32));
    }
    if (lane == 0) sm[g] = make_float2((mx - mn) * (1.0f / 15.0f), mn);
}

// ---------------- Pass 2: WMMA GEMM with fused dequant ---------------------
__global__ __launch_bounds__(256)
void bnb_gemm_wmma(const float* __restrict__ X, const float* __restrict__ W,
                   const float* __restrict__ bias, const float2* __restrict__ sm,
                   float* __restrict__ out) {
    __shared__ __bf16 As[BM][LDS_STRIDE];   // x tile,  [m][k]   36.9 KB
    __shared__ __bf16 Bs[BN][LDS_STRIDE];   // W tile,  [n][k]   18.4 KB

    const int t       = threadIdx.x;
    const int lane    = t & 31;
    const int wv      = t >> 5;       // 0..7 (8 waves)
    const int waveM   = wv >> 1;      // 0..3 -> 64-row slab
    const int waveN   = wv & 1;       // 0..1 -> 64-col slab
    const int ln      = lane & 15;
    const int halfsel = lane >> 4;    // 0/1

    const int m0 = blockIdx.y * BM;
    const int n0 = blockIdx.x * BN;

    const int lrow = t >> 4;          // 0..15 : row within a 16-row load pass
    const int kc   = (t & 15) * 4;    // 0..60 : float4 k-chunk

    v8f acc[4][4] = {};

    for (int k0 = 0; k0 < K_DIM; k0 += BK) {
        // ---- stage A (x, 256 rows) and B (dequantized W, 128 rows) ----
#pragma unroll
        for (int rb = 0; rb < 16; ++rb) {
            const int row = rb * 16 + lrow;

            const float4 xv = *(const float4*)(X + (size_t)(m0 + row) * K_DIM + k0 + kc);
            union { __bf16 h[4]; uint2 u; } pa;
            pa.h[0] = f2bf(xv.x); pa.h[1] = f2bf(xv.y);
            pa.h[2] = f2bf(xv.z); pa.h[3] = f2bf(xv.w);
            *(uint2*)&As[row][kc] = pa.u;

            if (rb < 8) {
                const int n = n0 + row;
                const float4 wq = *(const float4*)(W + (size_t)n * K_DIM + k0 + kc);
                const float2 s  = sm[n * GROUPS_PER_ROW + ((k0 + kc) >> 7)];
                union { __bf16 h[4]; uint2 u; } pb;
                pb.h[0] = f2bf(fmaf(wq.x, s.x, s.y));
                pb.h[1] = f2bf(fmaf(wq.y, s.x, s.y));
                pb.h[2] = f2bf(fmaf(wq.z, s.x, s.y));
                pb.h[3] = f2bf(fmaf(wq.w, s.x, s.y));
                *(uint2*)&Bs[row][kc] = pb.u;
                if (k0 + BK < K_DIM)
                    __builtin_prefetch(W + (size_t)n * K_DIM + k0 + BK + kc, 0, 1);
            }
            if (k0 + BK < K_DIM)  // overlap next tile fetch (global_prefetch_b8)
                __builtin_prefetch(X + (size_t)(m0 + row) * K_DIM + k0 + BK + kc, 0, 1);
        }
        __syncthreads();

        // ---- compute: 2 K-substeps of 32, 16 WMMAs each per wave ----
#pragma unroll
        for (int kk = 0; kk < BK; kk += 32) {
            union { v16bf v; v8bf h[2]; } a[4], b[4];
#pragma unroll
            for (int mi = 0; mi < 4; ++mi) {
                // A 16x32 bf16 layout: lane holds row M=lane%16;
                // lanes 0-15: K = kk+[0..7] & kk+16+[0..7]; lanes 16-31: +8
                const int r = waveM * 64 + mi * 16 + ln;
                a[mi].h[0] = *(const v8bf*)&As[r][kk + halfsel * 8];
                a[mi].h[1] = *(const v8bf*)&As[r][kk + 16 + halfsel * 8];
            }
#pragma unroll
            for (int ni = 0; ni < 4; ++ni) {
                // B 32x16 bf16 layout: lane holds col N=lane%16;
                // lanes 0-15: K = kk+[0..7] & kk+16+[0..7]; lanes 16-31: +8
                const int c = waveN * 64 + ni * 16 + ln;
                b[ni].h[0] = *(const v8bf*)&Bs[c][kk + halfsel * 8];
                b[ni].h[1] = *(const v8bf*)&Bs[c][kk + 16 + halfsel * 8];
            }
#pragma unroll
            for (int mi = 0; mi < 4; ++mi)
#pragma unroll
                for (int ni = 0; ni < 4; ++ni)
                    acc[mi][ni] = __builtin_amdgcn_wmma_f32_16x16x32_bf16(
                        false, a[mi].v, false, b[ni].v,
                        (short)0, acc[mi][ni], false, false);
        }
        __syncthreads();
    }

    // ---- epilogue: C layout VGPR r -> (M = r + 8*halfsel, N = lane%16) ----
#pragma unroll
    for (int ni = 0; ni < 4; ++ni) {
        const int col = n0 + waveN * 64 + ni * 16 + ln;
        const float bv = bias[col];
#pragma unroll
        for (int mi = 0; mi < 4; ++mi) {
            const int rbase = m0 + waveM * 64 + mi * 16 + halfsel * 8;
#pragma unroll
            for (int r = 0; r < 8; ++r)
                out[(size_t)(rbase + r) * N_DIM + col] = acc[mi][ni][r] + bv;
        }
    }
}

// ---------------------------------------------------------------------------
extern "C" void kernel_launch(void* const* d_in, const int* in_sizes, int n_in,
                              void* d_out, int out_size, void* d_ws, size_t ws_size,
                              hipStream_t stream) {
    (void)in_sizes; (void)n_in; (void)out_size; (void)ws_size;
    const float* X    = (const float*)d_in[0];
    const float* W    = (const float*)d_in[1];
    const float* bias = (const float*)d_in[2];
    float*  out = (float*)d_out;
    float2* sm  = (float2*)d_ws;   // NGROUPS * (scale, min) = ~2.8 MB scratch

    // Pass 1: one wave32 per quant group
    const long long threads1 = (long long)NGROUPS * 32;
    const int blocks1 = (int)((threads1 + 255) / 256);
    bnb_group_stats<<<blocks1, 256, 0, stream>>>(W, sm);

    // Pass 2: N-fastest block order so the 128-row weight tile stays L2-hot
    // across the 86 N-blocks of one M sweep (weight ~180MB vs 192MB L2).
    dim3 grid(N_DIM / BN, M_DIM / BM);   // (86, 32)
    bnb_gemm_wmma<<<grid, 256, 0, stream>>>(X, W, bias, sm, out);
}
